// GCN_91250875171104
// MI455X (gfx1250) — compile-verified, matrix operand order checked by reading, not run on hardware
//
#include <hip/hip_runtime.h>
#include <hip/hip_bf16.h>

typedef __attribute__((ext_vector_type(16))) _Float16 v16h;
typedef __attribute__((ext_vector_type(8)))  float    v8f;

// ---------------------------------------------------------------------------
// Kernel 1: deg[i] = 1 (self loop)
// ---------------------------------------------------------------------------
__global__ void gcn_init_deg(unsigned* __restrict__ deg, int n) {
    int i = blockIdx.x * blockDim.x + threadIdx.x;
    if (i < n) deg[i] = 1u;
}

// ---------------------------------------------------------------------------
// Kernel 2: deg[row[e]] += 1 for every edge
// ---------------------------------------------------------------------------
__global__ void gcn_count(const long long* __restrict__ ei,
                          unsigned* __restrict__ deg, long long E) {
    long long e = (long long)blockIdx.x * blockDim.x + threadIdx.x;
    if (e < E) atomicAdd(&deg[(int)ei[e]], 1u);
}

// ---------------------------------------------------------------------------
// Kernel 3: dis[i] = deg^-0.5 ; aggr[i][k] = x[i][k] / deg[i]  (self-loop msg)
// one thread per (node, feature); lane k==0 also writes dis
// ---------------------------------------------------------------------------
__global__ void gcn_prep(const unsigned* __restrict__ deg,
                         const float* __restrict__ x,
                         float* __restrict__ dis,
                         float* __restrict__ aggr, int n) {
    int t = blockIdx.x * blockDim.x + threadIdx.x;
    int i = t >> 5;
    int k = t & 31;
    if (i >= n) return;
    float d = (float)deg[i];
    if (k == 0) dis[i] = rsqrtf(d);
    aggr[t] = x[t] / d;
}

// ---------------------------------------------------------------------------
// Kernel 4: scatter-add messages. 8 lanes per edge, 4 features per lane:
// one float4 gather + 4 global_atomic_add_f32 per lane (128B per edge).
// aggr/x stay resident in the 192MB L2, so atomics resolve on-chip.
// ---------------------------------------------------------------------------
__global__ void gcn_scatter(const long long* __restrict__ ei,   // [2][E]
                            const float* __restrict__ ea,
                            const float* __restrict__ x,
                            const float* __restrict__ dis,
                            float* __restrict__ aggr, long long E) {
    long long t = (long long)blockIdx.x * blockDim.x + threadIdx.x;
    long long e = t >> 3;
    if (e >= E) return;
    int k0 = (int)(t & 7) * 4;
    long long r = ei[e];
    long long c = ei[E + e];
    float w = ea[e] * dis[(int)r] * dis[(int)c];
    const float4 xv = *(const float4*)(x + r * 32 + k0);
    float* ac = aggr + c * 32 + k0;
    atomicAdd(ac + 0, w * xv.x);
    atomicAdd(ac + 1, w * xv.y);
    atomicAdd(ac + 2, w * xv.z);
    atomicAdd(ac + 3, w * xv.w);
}

// ---------------------------------------------------------------------------
// Kernel 5: fused  out = sigmoid( tanh(aggr @ w1^T + b1) @ w2^T + b2 )
// One wave32 per 16-node tile. K=32 features == one V_WMMA_F32_16X16X32_F16.
// 4 WMMAs cover the 64 hidden units; second layer reduced with shfl_xor.
// ---------------------------------------------------------------------------
__global__ void gcn_wmma(const float* __restrict__ aggr,
                         const float* __restrict__ w1,   // [64][32]
                         const float* __restrict__ b1,   // [64]
                         const float* __restrict__ w2,   // [64]
                         const float* __restrict__ b2,   // [1]
                         float* __restrict__ out, int n_nodes) {
    const int wave = (blockIdx.x * blockDim.x + threadIdx.x) >> 5; // uniform per wave
    const int lane = threadIdx.x & 31;
    const int node_base = wave * 16;
    if (node_base >= n_nodes) return;          // wave-uniform exit: EXEC stays all-1

    const int m     = lane & 15;               // matrix row / col within tile
    const int kbase = (lane >> 4) * 8;         // lanes 0-15: K 0-7/16-23; 16-31: K 8-15/24-31
    int node = node_base + m;
    if (node >= n_nodes) node = n_nodes - 1;   // clamp loads; stores predicated later

    // ---- A tile: 16x32 f16 per ISA 7.12.2 layout ----
    v16h a;
    const float* arow = aggr + (long long)node * 32;
#pragma unroll
    for (int j = 0; j < 8; ++j) {
        a[j]     = (_Float16)arow[kbase + j];
        a[j + 8] = (_Float16)arow[kbase + 16 + j];
    }

    // ---- 4 output tiles of 16 hidden units each ----
    v8f acc[4];
#pragma unroll
    for (int t = 0; t < 4; ++t) {
        const int n = t * 16 + m;              // hidden-unit column this lane owns
        const float* wrow = w1 + n * 32;       // B[k][n] = w1[n][k]
        v16h b;
#pragma unroll
        for (int j = 0; j < 8; ++j) {
            b[j]     = (_Float16)wrow[kbase + j];
            b[j + 8] = (_Float16)wrow[kbase + 16 + j];
        }
        v8f c = {};
        acc[t] = __builtin_amdgcn_wmma_f32_16x16x32_f16(
            /*neg_a=*/false, a, /*neg_b=*/false, b,
            /*c_mod=*/(short)0, c, /*reuse_a=*/false, /*reuse_b=*/false);
    }

    // ---- layer 2: dot_n( tanh(acc + b1[n]), w2[n] ) ----
    // lane holds hidden unit n = t*16 + m; reg r maps to node row r + 8*(lane>=16)
    float dot[8];
#pragma unroll
    for (int r = 0; r < 8; ++r) dot[r] = 0.0f;
#pragma unroll
    for (int t = 0; t < 4; ++t) {
        const int n  = t * 16 + m;
        const float bn = b1[n];
        const float wn = w2[n];
#pragma unroll
        for (int r = 0; r < 8; ++r)
            dot[r] += tanhf(acc[t][r] + bn) * wn;
    }
    // butterfly-sum across the 16 lanes of each half (each half owns 8 node rows)
#pragma unroll
    for (int mask = 1; mask < 16; mask <<= 1) {
#pragma unroll
        for (int r = 0; r < 8; ++r)
            dot[r] += __shfl_xor(dot[r], mask, 32);
    }

    if (m == 0) {
        const float bb = b2[0];
        const int rowbase = node_base + (lane >> 4) * 8;
#pragma unroll
        for (int r = 0; r < 8; ++r) {
            const int nd = rowbase + r;
            if (nd < n_nodes) {
                const float v = dot[r] + bb;
                out[nd] = 1.0f / (1.0f + expf(-v));
            }
        }
    }
}

// ---------------------------------------------------------------------------
// Host-side launcher
// ---------------------------------------------------------------------------
extern "C" void kernel_launch(void* const* d_in, const int* in_sizes, int n_in,
                              void* d_out, int out_size, void* d_ws, size_t ws_size,
                              hipStream_t stream) {
    const float*     x  = (const float*)d_in[0];                 // [N][32] f32
    const long long* ei = (const long long*)d_in[1];             // [2][E] i64
    const float*     ea = (const float*)d_in[2];                 // [E]    f32
    const float*     w1 = (const float*)d_in[3];                 // [64][32]
    const float*     b1 = (const float*)d_in[4];                 // [64]
    const float*     w2 = (const float*)d_in[5];                 // [1][64]
    const float*     b2 = (const float*)d_in[6];                 // [1]
    float* out = (float*)d_out;

    const int       N = in_sizes[0] / 32;
    const long long E = (long long)in_sizes[2];

    // workspace carve-out: aggr[N*32] f32 | dis[N] f32 | deg[N] u32
    char* ws = (char*)d_ws;
    float*    aggr = (float*)ws;
    float*    dis  = (float*)(ws + (size_t)N * 32 * sizeof(float));
    unsigned* deg  = (unsigned*)(ws + (size_t)N * 32 * sizeof(float)
                                    + (size_t)N * sizeof(float));
    (void)n_in; (void)out_size; (void)ws_size;

    const int B = 256;
    gcn_init_deg<<<(N + B - 1) / B, B, 0, stream>>>(deg, N);
    gcn_count<<<(unsigned)((E + B - 1) / B), B, 0, stream>>>(ei, deg, E);
    gcn_prep<<<(unsigned)(((long long)N * 32 + B - 1) / B), B, 0, stream>>>(deg, x, dis, aggr, N);
    gcn_scatter<<<(unsigned)((E * 8 + B - 1) / B), B, 0, stream>>>(ei, ea, x, dis, aggr, E);

    const int tiles  = (N + 15) / 16;          // one wave32 per tile
    const int wavesPerBlock = 8;               // 256 threads
    const int blocks = (tiles + wavesPerBlock - 1) / wavesPerBlock;
    gcn_wmma<<<blocks, wavesPerBlock * 32, 0, stream>>>(aggr, w1, b1, w2, b2, out, N);
}
